// ISAR_render_55619826483532
// MI455X (gfx1250) — matrix operand order repaired and checked by reading, not compiled, not run on hardware
//
#include <hip/hip_runtime.h>
#include <stdint.h>

typedef __attribute__((ext_vector_type(16))) _Float16 v16h;
typedef __attribute__((ext_vector_type(8)))  _Float16 v8h;
typedef __attribute__((ext_vector_type(2)))  _Float16 v2h;
typedef __attribute__((ext_vector_type(8)))  float    v8f;
typedef __attribute__((ext_vector_type(4)))  uint32_t u32x4;
typedef __attribute__((ext_vector_type(8)))  uint32_t u32x8;

#define GRID_N 100
#define NPTS   8000
#define B_SZ   4
#define RPAD   112                 // 7*16 padded grid lines
#define KS     800                 // K-slab per TDM stage
#define NSLAB  (NPTS / KS)         // 10
#define KCH    (KS / 32)           // 25 K-chunks of 32 per slab

// ---------------------------------------------------------------------------
// Kernel 0: per-point scalars (range, doppler, amp) -> workspace f32 arrays
// ---------------------------------------------------------------------------
__global__ void isar_scalars(const float* __restrict__ pts,
                             const float* __restrict__ nrm,
                             const float* __restrict__ los_p,
                             const float* __restrict__ spin_p,
                             const float* __restrict__ omega_p,
                             float* __restrict__ dop,
                             float* __restrict__ rng,
                             float* __restrict__ amp)
{
    const int i = blockIdx.x * blockDim.x + threadIdx.x;
    if (i >= B_SZ * NPTS) return;
    const float LAMBDA1 = 299792458.0f / 9.7e9f;
    const float om = omega_p[0];
    const float lx = los_p[0],  ly = los_p[1],  lz = los_p[2];
    const float sx = spin_p[0], sy = spin_p[1], sz = spin_p[2];
    const int g = i * 3;
    const float px = pts[g], py = pts[g + 1], pz = pts[g + 2];
    const float nx = nrm[g], ny = nrm[g + 1], nz = nrm[g + 2];
    const float vx = sy * pz - sz * py;
    const float vy = sz * px - sx * pz;
    const float vz = sx * py - sy * px;
    const float vr = om * (lx * vx + ly * vy + lz * vz);
    dop[i] = -2.0f * vr / LAMBDA1;
    rng[i] = lx * px + ly * py + lz * pz;
    const float a = -4.0f * (lx * nx + ly * ny + lz * nz);
    amp[i] = fminf(fmaxf(a, 0.0f), 1.0f);
}

// ---------------------------------------------------------------------------
// Kernel 1: Gaussian factor matrices (f16), each exp computed exactly once.
//   Aw[b][r][k] = amp * exp(-0.5((dop-mu_d(r))/dres)^2)   r in [0,112)
//   Bw[b][r][k] =       exp(-0.5((rng-mu_r(r))/rres)^2)
// ---------------------------------------------------------------------------
__global__ void isar_factors(const float* __restrict__ dop,
                             const float* __restrict__ rng,
                             const float* __restrict__ amp,
                             const float* __restrict__ omega_p,
                             _Float16* __restrict__ Aw,
                             _Float16* __restrict__ Bw)
{
    const int k0 = blockIdx.x * 512 + threadIdx.x * 2;   // 2 points per thread
    if (k0 >= NPTS) return;
    const int r  = blockIdx.y;            // grid line 0..111
    const int z  = blockIdx.z;            // 0..7 : batch*2 + isB
    const int b  = z >> 1;
    const int isB = z & 1;

    const float LAMBDA1 = 299792458.0f / 9.7e9f;
    const float om = omega_p[0];
    const float inv_dres = (2.0f * om * 160.0f) / LAMBDA1;             // 1/doppler_res
    const float inv_rres = 1.0f / (299792458.0f / 4.8e9f * 2.5f);      // 1/RANGE_RES

    const int base = b * NPTS + k0;
    float v0, v1;
    if (!isB) {
        const float mu = -3.0f + (6.0f / 99.0f) * (float)r;
        const float t0 = (dop[base]     - mu) * inv_dres;
        const float t1 = (dop[base + 1] - mu) * inv_dres;
        v0 = amp[base]     * __expf(-0.5f * t0 * t0);
        v1 = amp[base + 1] * __expf(-0.5f * t1 * t1);
    } else {
        const float mu = -10.0f + (20.0f / 99.0f) * (float)r;
        const float t0 = (rng[base]     - mu) * inv_rres;
        const float t1 = (rng[base + 1] - mu) * inv_rres;
        v0 = __expf(-0.5f * t0 * t0);
        v1 = __expf(-0.5f * t1 * t1);
    }
    _Float16* dst = (isB ? Bw : Aw) + ((size_t)(b * RPAD + r) * NPTS + k0);
    v2h pk = { (_Float16)v0, (_Float16)v1 };
    *(v2h*)dst = pk;
}

// ---------------------------------------------------------------------------
// Kernel 2: tiled WMMA GEMM  C[b] = A[b] (112x8000) x B[b]^T (8000x112),
// operand tiles staged LDS-side by the Tensor Data Mover (tensor_load_to_lds).
// ---------------------------------------------------------------------------
__launch_bounds__(256, 1)
__global__ void isar_gemm_wmma(const _Float16* __restrict__ Aw,
                               const _Float16* __restrict__ Bw,
                               float* __restrict__ out)
{
    extern __shared__ __align__(16) _Float16 smem[]; // dynamic LDS, offset 0
    _Float16* sA = smem;                  // [16][KS] f16 tile of A
    _Float16* sB = smem + 16 * KS;        // [16][KS] f16 tile of B^T
    float*    sC = (float*)(smem + 32 * KS); // [256] reduction tile

    const int tid  = threadIdx.x;
    const int lane = tid & 31;
    const int wave = tid >> 5;
    const int half = lane >> 4;           // 0: lanes 0-15, 1: lanes 16-31
    const int mloc = lane & 15;           // A row / B col within tile
    const int tj = blockIdx.x, ti = blockIdx.y, b = blockIdx.z;

    sC[tid] = 0.0f;
    v8f acc = {};

    const uint32_t ldsA = 0u;
    const uint32_t ldsB = (uint32_t)(16 * KS * 2);

    for (int s = 0; s < NSLAB; ++s) {
        __syncthreads();                  // previous slab fully consumed
        if (wave == 0) {
            // --- Tensor DMA descriptors (CDNA5 ISA D#) ---
            const uint64_t ga = (uint64_t)(uintptr_t)
                (Aw + ((size_t)(b * RPAD + ti * 16) * NPTS + s * KS));
            const uint64_t gb = (uint64_t)(uintptr_t)
                (Bw + ((size_t)(b * RPAD + tj * 16) * NPTS + s * KS));
            // group0: count=1 | lds_addr | global_addr[56:0] | type=2
            u32x4 g0a = { 1u, ldsA, (uint32_t)ga,
                          ((uint32_t)(ga >> 32) & 0x01FFFFFFu) | 0x80000000u };
            u32x4 g0b = { 1u, ldsB, (uint32_t)gb,
                          ((uint32_t)(gb >> 32) & 0x01FFFFFFu) | 0x80000000u };
            // group1: data_size=2B; tensor 8000x112; tile KSx16; stride0=8000
            u32x8 g1 = { 0x00010000u,
                         ((uint32_t)NPTS & 0xFFFFu) << 16,
                         ((uint32_t)NPTS >> 16) | ((uint32_t)RPAD << 16),
                         ((uint32_t)KS << 16),
                         16u,
                         (uint32_t)NPTS,
                         0u, 0u };
            asm volatile("tensor_load_to_lds %0, %1" :: "s"(g0a), "s"(g1) : "memory");
            asm volatile("tensor_load_to_lds %0, %1" :: "s"(g0b), "s"(g1) : "memory");
            __builtin_amdgcn_s_wait_tensorcnt(0);
        }
        __syncthreads();                  // tile visible to all waves

        // 8 waves partition the 25 K-chunks of this slab (EXEC all-ones)
        for (int c = wave; c < KCH; c += 8) {
            const int kc = c * 32;
            // A fragment (ISA 7.12.2): lane holds row M=mloc, K runs
            // [kc+8h, +8) and [kc+16+8h, +8)  -> two 16B LDS loads
            const _Float16* ap = sA + mloc * KS + kc + 8 * half;
            v8h alo = *(const v8h*)(ap);
            v8h ahi = *(const v8h*)(ap + 16);
            // B fragment: lane holds col N=mloc, K runs
            // [kc+8h, +8) and [kc+16+8h, +8)  -> two 16B LDS loads
            const _Float16* bp = sB + mloc * KS + kc + 8 * half;
            v8h blo = *(const v8h*)(bp);
            v8h bhi = *(const v8h*)(bp + 16);
            v16h af = __builtin_shufflevector(alo, ahi,
                        0,1,2,3,4,5,6,7,8,9,10,11,12,13,14,15);
            v16h bf = __builtin_shufflevector(blo, bhi,
                        0,1,2,3,4,5,6,7,8,9,10,11,12,13,14,15);
            acc = __builtin_amdgcn_wmma_f32_16x16x32_f16(
                false, af, false, bf, (short)0, acc, false, false);
        }
    }

    __syncthreads();
    // cross-wave reduction of partial 16x16 tiles (C layout: M=r+8*half, N=mloc)
    #pragma unroll
    for (int r = 0; r < 8; ++r)
        atomicAdd(&sC[(r + 8 * half) * 16 + mloc], acc[r]);
    __syncthreads();

    const int gi = ti * 16 + (tid >> 4);
    const int gj = tj * 16 + (tid & 15);
    if (gi < GRID_N && gj < GRID_N)
        out[(b * GRID_N + gi) * GRID_N + gj] = sC[tid];
}

// ---------------------------------------------------------------------------
// Fallback: fused recompute kernel (round-1, compile-proven) if ws too small.
// ---------------------------------------------------------------------------
__launch_bounds__(256, 1)
__global__ void isar_render_fused(const float* __restrict__ pts,
                                  const float* __restrict__ nrm,
                                  const float* __restrict__ los_p,
                                  const float* __restrict__ spin_p,
                                  const float* __restrict__ omega_p,
                                  float* __restrict__ out)
{
    __shared__ float s_range[1600];
    __shared__ float s_dop[1600];
    __shared__ float s_amp[1600];
    __shared__ float s_tile[256];

    const int tid  = threadIdx.x;
    const int lane = tid & 31;
    const int wave = tid >> 5;
    const int half = lane >> 4;
    const int mloc = lane & 15;
    const int tj = blockIdx.x, ti = blockIdx.y, b = blockIdx.z;

    const float LAMBDA1 = 299792458.0f / 9.7e9f;
    const float om = omega_p[0];
    const float lx = los_p[0],  ly = los_p[1],  lz = los_p[2];
    const float sx = spin_p[0], sy = spin_p[1], sz = spin_p[2];
    const float inv_rres = 1.0f / (299792458.0f / 4.8e9f * 2.5f);
    const float inv_dres = (2.0f * om * 160.0f) / LAMBDA1;
    const float mu_dop = -3.0f  + (6.0f / 99.0f)  * (float)(ti * 16 + mloc);
    const float mu_rng = -10.0f + (20.0f / 99.0f) * (float)(tj * 16 + mloc);

    s_tile[tid] = 0.0f;
    v8f acc = {};

    for (int st = 0; st < 5; ++st) {
        __syncthreads();
        for (int p = tid; p < 1600; p += 256) {
            const int g = (b * NPTS + st * 1600 + p) * 3;
            const float px = pts[g], py = pts[g + 1], pz = pts[g + 2];
            const float nx = nrm[g], ny = nrm[g + 1], nz = nrm[g + 2];
            const float vx = sy * pz - sz * py;
            const float vy = sz * px - sx * pz;
            const float vz = sx * py - sy * px;
            const float vr = om * (lx * vx + ly * vy + lz * vz);
            s_dop[p]   = -2.0f * vr / LAMBDA1;
            s_range[p] = lx * px + ly * py + lz * pz;
            const float a = -4.0f * (lx * nx + ly * ny + lz * nz);
            s_amp[p]   = fminf(fmaxf(a, 0.0f), 1.0f);
        }
        __syncthreads();
        for (int c = wave; c < 50; c += 8) {
            const int kb = c * 32;
            v16h af, bf;
            #pragma unroll
            for (int v = 0; v < 8; ++v) {
                #pragma unroll
                for (int e = 0; e < 2; ++e) {
                    const int ka = 2 * v + e + 8 * half + ((v >= 4) ? 8 : 0);
                    const int pa = kb + ka;
                    const float td = (s_dop[pa] - mu_dop) * inv_dres;
                    af[2 * v + e] = (_Float16)(s_amp[pa] * __expf(-0.5f * td * td));
                    const int pb = kb + 2 * v + e + 8 * half + ((v >= 4) ? 8 : 0);
                    const float tr = (s_range[pb] - mu_rng) * inv_rres;
                    bf[2 * v + e] = (_Float16)__expf(-0.5f * tr * tr);
                }
            }
            acc = __builtin_amdgcn_wmma_f32_16x16x32_f16(
                false, af, false, bf, (short)0, acc, false, false);
        }
    }
    __syncthreads();
    #pragma unroll
    for (int r = 0; r < 8; ++r)
        atomicAdd(&s_tile[(r + 8 * half) * 16 + mloc], acc[r]);
    __syncthreads();
    const int gi = ti * 16 + (tid >> 4);
    const int gj = tj * 16 + (tid & 15);
    if (gi < GRID_N && gj < GRID_N)
        out[(b * GRID_N + gi) * GRID_N + gj] = s_tile[tid];
}

// ---------------------------------------------------------------------------
extern "C" void kernel_launch(void* const* d_in, const int* in_sizes, int n_in,
                              void* d_out, int out_size, void* d_ws, size_t ws_size,
                              hipStream_t stream) {
    const float* pts  = (const float*)d_in[0];  // [4,8000,3]
    const float* nrm  = (const float*)d_in[1];  // [4,8000,3]
    const float* los  = (const float*)d_in[2];  // [1,3]
    const float* spin = (const float*)d_in[3];  // [1,3]
    const float* omg  = (const float*)d_in[4];  // [1,1]
    float* out = (float*)d_out;                 // [4,100,100]

    const size_t n_sc   = (size_t)B_SZ * NPTS;           // 32000
    const size_t n_fac  = (size_t)B_SZ * RPAD * NPTS;    // 3,584,000
    const size_t need   = 3 * n_sc * sizeof(float) + 2 * n_fac * sizeof(_Float16);

    if (ws_size >= need) {
        float* dop = (float*)d_ws;
        float* rng = dop + n_sc;
        float* amp = rng + n_sc;
        _Float16* Aw = (_Float16*)(amp + n_sc);
        _Float16* Bw = Aw + n_fac;

        isar_scalars<<<(B_SZ * NPTS + 255) / 256, 256, 0, stream>>>(
            pts, nrm, los, spin, omg, dop, rng, amp);

        dim3 gf(16, RPAD, 2 * B_SZ);   // k-blocks x grid-lines x (batch*2)
        isar_factors<<<gf, 256, 0, stream>>>(dop, rng, amp, omg, Aw, Bw);

        dim3 gg(7, 7, B_SZ);
        const size_t shbytes = 32 * KS * sizeof(_Float16) + 256 * sizeof(float);
        isar_gemm_wmma<<<gg, 256, shbytes, stream>>>(Aw, Bw, out);
    } else {
        dim3 grid(7, 7, B_SZ);
        isar_render_fused<<<grid, 256, 0, stream>>>(pts, nrm, los, spin, omg, out);
    }
}
